// Attention_79010218377454
// MI455X (gfx1250) — matrix operand order repaired and checked by reading, not compiled, not run on hardware
//
#include <hip/hip_runtime.h>
#include <math.h>

// ---------------------------------------------------------------------------
// MHA forward for MI455X (gfx1250): bf16 WMMA + TDM async tile loads.
// ---------------------------------------------------------------------------

typedef __bf16 bf16;
typedef __bf16 v16bf __attribute__((ext_vector_type(16)));
typedef __bf16 v8bf  __attribute__((ext_vector_type(8)));
typedef float  v8f   __attribute__((ext_vector_type(8)));
typedef float  v4f   __attribute__((ext_vector_type(4)));
typedef unsigned int u32x4 __attribute__((ext_vector_type(4)));
typedef int          i32x4 __attribute__((ext_vector_type(4)));
typedef int          i32x8 __attribute__((ext_vector_type(8)));

#define D_MODEL 1024
#define SEQ     2048
#define BATCH   4
#define NHEAD   16
#define HDIM    64
#define MTOT    (BATCH * SEQ)   // 8192 rows

// GEMM tiling: 128x64 tile per 256-thread (8-wave) workgroup, K step 32.
#define BM 128
#define BN 64
#define BK 32
#define LDT 40    // padded LDS stride (bf16 elems): 80B, multiple of 16B
#define LDQ 72    // padded stride for 64-wide attention tiles: 144B, mult of 16B

__device__ inline v8f zero8() {
  v8f z;
#pragma unroll
  for (int i = 0; i < 8; ++i) z[i] = 0.0f;
  return z;
}

__device__ inline v8f wmma_bf16(v16bf a, v16bf b, v8f c) {
  // D = A(16x32 bf16) * B(32x16 bf16) + C(16x16 f32)
  return __builtin_amdgcn_wmma_f32_16x16x32_bf16(
      /*neg_a=*/false, a, /*neg_b=*/false, b,
      /*c_mod=*/(short)0, c, /*reuse_a=*/false, /*reuse_b=*/false);
}

// A fragment (16x32, 16-bit): lane half 0 holds K 0..7 & 16..23, half 1 holds
// K 8..15 & 24..31 (ISA 7.12.2). rowbase points at element [m][0] of a
// K-contiguous LDS row; two aligned 16B loads.
__device__ inline v16bf load_a_frag(const bf16* rowbase, int lane) {
  const int kb = ((lane >> 4) << 3);           // 0 or 8
  v8bf lo = *(const v8bf*)(rowbase + kb);      // K = kb .. kb+7
  v8bf hi = *(const v8bf*)(rowbase + 16 + kb); // K = 16+kb .. 16+kb+7
  v16bf r;
#pragma unroll
  for (int i = 0; i < 8; ++i) { r[i] = lo[i]; r[i + 8] = hi[i]; }
  return r;
}

// B fragment (32x16, 16-bit): lane half 0 holds K 0..7 & 16..23, half 1 K 8..15 & 24..31.
// rowbase points at element [n][0] of a K-contiguous (transposed) LDS row.
__device__ inline v16bf load_b_frag(const bf16* rowbase, int lane) {
  const int kb = ((lane >> 4) << 3);           // 0 or 8
  v8bf lo = *(const v8bf*)(rowbase + kb);
  v8bf hi = *(const v8bf*)(rowbase + 16 + kb);
  v16bf r;
#pragma unroll
  for (int i = 0; i < 8; ++i) { r[i] = lo[i]; r[i + 8] = hi[i]; }
  return r;
}

// ---------------------------------------------------------------------------
// Tensor Data Mover: async 2D bf16 tile load global -> LDS (ISA ch. 7/8 of
// CDNA5 ISA). tile_w elements per row (contiguous in memory),
// tile_h rows, row stride in elements. LDS destination gets +4 dwords of
// padding after every 32 dwords (=64 bf16 row) -> padded stride LDQ=72.
// Tensor dims are oversized so the DMA never clamps (tiles are in-bounds by
// construction). Tracked with TENSORcnt; completion = data visible in LDS.
// ---------------------------------------------------------------------------
__device__ inline void tdm_load_tile_2d(unsigned lds_byte_addr,
                                        const void* gaddr,
                                        unsigned tile_w, unsigned tile_h,
                                        unsigned long long row_stride_elems) {
  const unsigned long long ga = (unsigned long long)(size_t)gaddr;

  u32x4 g0;
  g0[0] = 1u;                                          // count=1, user D#
  g0[1] = lds_byte_addr;                               // lds_addr [63:32]
  g0[2] = (unsigned)ga;                                // global_addr lo
  g0[3] = (unsigned)((ga >> 32) & 0x01FFFFFFu)         // global_addr hi
          | (2u << 30);                                // type = 2 ("image")

  const unsigned td0 = 1u << 20;                       // oversized tensor dims
  const unsigned td1 = 1u << 20;
  i32x8 g1;
  g1[0] = (int)((1u << 16)       // data_size = 1 -> 2 bytes
                | (1u << 20)     // pad_enable
                | (4u << 22)     // pad_interval: 32 dwords (128B = 64 bf16)
                | (3u << 25));   // pad_amount: 4 dwords (16B = 8 bf16)
  g1[1] = (int)((td0 & 0xFFFFu) << 16);                // tensor_dim0 [79:48]
  g1[2] = (int)((td0 >> 16) | ((td1 & 0xFFFFu) << 16));// dim0 hi | dim1 lo
  g1[3] = (int)((td1 >> 16) | (tile_w << 16));         // dim1 hi | tile_dim0
  g1[4] = (int)(tile_h & 0xFFFFu);                     // tile_dim1, tile_dim2=0
  g1[5] = (int)(unsigned)(row_stride_elems & 0xFFFFFFFFull); // dim0_stride lo
  g1[6] = (int)(unsigned)((row_stride_elems >> 32) & 0xFFFFull);
  g1[7] = 0;                                           // dim1_stride (2D: n/a)

  i32x4 z4;
#pragma unroll
  for (int i = 0; i < 4; ++i) z4[i] = 0;
#if defined(__clang_major__) && __clang_major__ >= 23
  i32x8 z8;
#pragma unroll
  for (int i = 0; i < 8; ++i) z8[i] = 0;
  __builtin_amdgcn_tensor_load_to_lds(g0, g1, z4, z4, z8, 0);
#else
  __builtin_amdgcn_tensor_load_to_lds(g0, g1, z4, z4, 0);
#endif
}

// ---------------------------------------------------------------------------
// Kernel 1: projection GEMM. Y = (X @ W + b) * scale, stored bf16.
// vtranspose==0: head-split [B, H, S, HD] (for Q, K).
// vtranspose==1: head-split transposed [B, H, HD, S] (for V; makes the V
//                attention tile a plain 2D TDM tile landing in [d][kv]).
// ---------------------------------------------------------------------------
__global__ __launch_bounds__(256)
void proj_gemm_kernel(const float* __restrict__ X, const float* __restrict__ W,
                      const float* __restrict__ bias, bf16* __restrict__ Yheads,
                      float scale, int vtranspose) {
  __shared__ __align__(16) bf16 As[BM * LDT];  // [m][k], K-contiguous
  __shared__ __align__(16) bf16 Bs[BN * LDT];  // [n][k], transposed

  const int tid = threadIdx.x;
  const int lane = tid & 31;
  const int wv = tid >> 5;              // 8 waves
  const int m0 = blockIdx.y * BM;
  const int n0 = blockIdx.x * BN;
  const int half = lane >> 4;
  const int l16 = lane & 15;

  v8f acc[4];
#pragma unroll
  for (int t = 0; t < 4; ++t) acc[t] = zero8();

  for (int k0 = 0; k0 < D_MODEL; k0 += BK) {
    // A tile: 128x32 f32 -> bf16. 1024 float4 loads, 4 per thread.
#pragma unroll
    for (int i = 0; i < 4; ++i) {
      const int f = tid + 256 * i;
      const int r = f >> 3;
      const int c = (f & 7) << 2;
      v4f val = *(const v4f*)(X + (size_t)(m0 + r) * D_MODEL + k0 + c);
      bf16* dst = As + r * LDT + c;
      dst[0] = (bf16)val.x; dst[1] = (bf16)val.y;
      dst[2] = (bf16)val.z; dst[3] = (bf16)val.w;
    }
    // B tile: 32x64 f32 -> bf16 transposed into [n][k]. 512 float4, 2/thread.
#pragma unroll
    for (int i = 0; i < 2; ++i) {
      const int f = tid + 256 * i;
      const int kr = f >> 4;
      const int nc = (f & 15) << 2;
      v4f val = *(const v4f*)(W + (size_t)(k0 + kr) * D_MODEL + n0 + nc);
      Bs[(nc + 0) * LDT + kr] = (bf16)val.x;
      Bs[(nc + 1) * LDT + kr] = (bf16)val.y;
      Bs[(nc + 2) * LDT + kr] = (bf16)val.z;
      Bs[(nc + 3) * LDT + kr] = (bf16)val.w;
    }
    __syncthreads();

    v16bf a = load_a_frag(As + (16 * wv + l16) * LDT, lane);
#pragma unroll
    for (int t = 0; t < 4; ++t) {
      v16bf b = load_b_frag(Bs + (16 * t + l16) * LDT, lane);
      acc[t] = wmma_bf16(a, b, acc[t]);
    }
    __syncthreads();
  }

  // Store: C layout — VGPR r holds row r (+8 for lane half 1), col = lane&15.
#pragma unroll
  for (int t = 0; t < 4; ++t) {
    const int n = n0 + 16 * t + l16;
    const float bv = bias[n];
    const int h = n >> 6;
    const int hd = n & 63;
#pragma unroll
    for (int r = 0; r < 8; ++r) {
      const int m = m0 + 16 * wv + r + half * 8;
      const int b = m >> 11;       // / SEQ
      const int s = m & 2047;      // % SEQ
      const float y = (acc[t][r] + bv) * scale;
      const size_t idx =
          vtranspose ? ((size_t)(b * NHEAD + h) * HDIM + hd) * SEQ + s
                     : ((size_t)(b * NHEAD + h) * SEQ + s) * HDIM + hd;
      Yheads[idx] = (bf16)y;
    }
  }
}

// ---------------------------------------------------------------------------
// Kernel 2: flash attention. One WG (4 waves, 128 thr) per (b, h, 64-q block).
// Qh already scaled by 1/sqrt(HD). Kh: [B,H,S,HD]; Vh: [B,H,HD,S] transposed.
// K/V tiles double-buffered via TDM async loads. Writes ctx bf16 as [B,S,D].
// ---------------------------------------------------------------------------
__global__ __launch_bounds__(128)
void attn_kernel(const bf16* __restrict__ Qh, const bf16* __restrict__ Kh,
                 const bf16* __restrict__ Vh, bf16* __restrict__ Ctx) {
  __shared__ __align__(16) bf16 Qs[64 * LDQ];        // [m][d]
  __shared__ __align__(16) bf16 Ks[2][64 * LDQ];     // [kv][d] double-buffered
  __shared__ __align__(16) bf16 Vs[2][64 * LDQ];     // [d][kv] double-buffered
  __shared__ __align__(16) bf16 Ps[4 * 16 * LDQ];    // per-wave P staging

  const int tid = threadIdx.x;
  const int lane = tid & 31;
  const int wv = tid >> 5;                           // 4 waves
  const int half = lane >> 4;
  const int l16 = lane & 15;
  const int mblk = blockIdx.x;
  const int h = blockIdx.y;
  const int b = blockIdx.z;
  const int NB = SEQ / 64;

  const size_t headbase = (size_t)(b * NHEAD + h) * SEQ * HDIM;
  const bf16* Qg = Qh + headbase + (size_t)mblk * 64 * HDIM;
  const bf16* Kg = Kh + headbase;   // rows s, stride HDIM
  const bf16* Vg = Vh + headbase;   // rows d, stride SEQ (transposed layout)

  // Kick off Q tile + first K/V tiles on the Tensor Data Mover (wave 0 only;
  // TDM ignores EXEC, descriptors are wave-uniform, one issue per wave).
  if (wv == 0) {
    tdm_load_tile_2d((unsigned)(size_t)(void*)Qs, Qg, 64, 64, HDIM);
    tdm_load_tile_2d((unsigned)(size_t)(void*)Ks[0], Kg, 64, 64, HDIM);
    tdm_load_tile_2d((unsigned)(size_t)(void*)Vs[0], Vg, 64, 64, SEQ);
  }

  v8f o[4];
#pragma unroll
  for (int t = 0; t < 4; ++t) o[t] = zero8();
  float mstat[8], lstat[8];
#pragma unroll
  for (int r = 0; r < 8; ++r) { mstat[r] = -1e30f; lstat[r] = 0.0f; }

  for (int j = 0; j < NB; ++j) {
    // Prefetch next K/V tiles into the other buffer, then wait until the
    // current tile's DMA (issued in-order before them) has completed.
    if (wv == 0) {
      if (j + 1 < NB) {
        const int nb = (j + 1) & 1;
        tdm_load_tile_2d((unsigned)(size_t)(void*)Ks[nb],
                         Kg + (size_t)(j + 1) * 64 * HDIM, 64, 64, HDIM);
        tdm_load_tile_2d((unsigned)(size_t)(void*)Vs[nb],
                         Vg + (size_t)(j + 1) * 64, 64, 64, SEQ);
        __builtin_amdgcn_s_wait_tensorcnt(2);
      } else {
        __builtin_amdgcn_s_wait_tensorcnt(0);
      }
    }
    __syncthreads();  // current tile resident for all waves

    const bf16* Ksb = Ks[j & 1];
    const bf16* Vsb = Vs[j & 1];

    // S = Q @ K^T for this wave's 16 rows x 64 kv cols.
    v8f sc[4];
#pragma unroll
    for (int t = 0; t < 4; ++t) sc[t] = zero8();
#pragma unroll
    for (int kk = 0; kk < 2; ++kk) {
      v16bf a = load_a_frag(Qs + (16 * wv + l16) * LDQ + kk * 32, lane);
#pragma unroll
      for (int t = 0; t < 4; ++t) {
        v16bf bb = load_b_frag(Ksb + (16 * t + l16) * LDQ + kk * 32, lane);
        sc[t] = wmma_bf16(a, bb, sc[t]);
      }
    }

    // Online softmax. Row r lives in the 16 lanes of this lane's half-wave,
    // so xor masks 1,2,4,8 reduce exactly over one row.
#pragma unroll
    for (int r = 0; r < 8; ++r) {
      float rowmax = fmaxf(fmaxf(sc[0][r], sc[1][r]), fmaxf(sc[2][r], sc[3][r]));
#pragma unroll
      for (int d = 8; d >= 1; d >>= 1)
        rowmax = fmaxf(rowmax, __shfl_xor(rowmax, d, 32));
      const float mnew = fmaxf(mstat[r], rowmax);
      const float alpha = __expf(mstat[r] - mnew);
      mstat[r] = mnew;
      float psum = 0.0f;
#pragma unroll
      for (int t = 0; t < 4; ++t) {
        const float p = __expf(sc[t][r] - mnew);
        sc[t][r] = p;
        psum += p;
      }
#pragma unroll
      for (int d = 8; d >= 1; d >>= 1) psum += __shfl_xor(psum, d, 32);
      lstat[r] = lstat[r] * alpha + psum;
#pragma unroll
      for (int t = 0; t < 4; ++t) o[t][r] *= alpha;
    }

    // Stage P (C layout -> A layout) through this wave's private LDS region.
    bf16* Pw = Ps + wv * 16 * LDQ;
#pragma unroll
    for (int t = 0; t < 4; ++t)
#pragma unroll
      for (int r = 0; r < 8; ++r)
        Pw[(r + half * 8) * LDQ + 16 * t + l16] = (bf16)sc[t][r];
    asm volatile("s_wait_dscnt 0" ::: "memory");  // P stores visible to frag loads

    // O += P @ V
#pragma unroll
    for (int kk = 0; kk < 2; ++kk) {
      v16bf a = load_a_frag(Pw + l16 * LDQ + kk * 32, lane);
#pragma unroll
      for (int t = 0; t < 4; ++t) {
        v16bf bb = load_b_frag(Vsb + (16 * t + l16) * LDQ + kk * 32, lane);
        o[t] = wmma_bf16(a, bb, o[t]);
      }
    }
    __syncthreads();  // all waves done with buffer (j&1) before its reuse
  }

  // Normalize and store ctx (merge heads back to [B, S, D]).
#pragma unroll
  for (int r = 0; r < 8; ++r) {
    const float inv = 1.0f / lstat[r];
    const int qrow = mblk * 64 + 16 * wv + r + half * 8;
    const size_t rowoff = ((size_t)b * SEQ + qrow) * D_MODEL + h * HDIM;
#pragma unroll
    for (int t = 0; t < 4; ++t)
      Ctx[rowoff + 16 * t + l16] = (bf16)(o[t][r] * inv);
  }
}

// ---------------------------------------------------------------------------
// Kernel 3: output projection. out = ctx(bf16) @ Wo + bo, f32 out [MTOT, D].
// ---------------------------------------------------------------------------
__global__ __launch_bounds__(256)
void out_gemm_kernel(const bf16* __restrict__ Ctx, const float* __restrict__ W,
                     const float* __restrict__ bias, float* __restrict__ Y) {
  __shared__ __align__(16) bf16 As[BM * LDT];
  __shared__ __align__(16) bf16 Bs[BN * LDT];

  const int tid = threadIdx.x;
  const int lane = tid & 31;
  const int wv = tid >> 5;
  const int m0 = blockIdx.y * BM;
  const int n0 = blockIdx.x * BN;
  const int half = lane >> 4;
  const int l16 = lane & 15;

  v8f acc[4];
#pragma unroll
  for (int t = 0; t < 4; ++t) acc[t] = zero8();

  for (int k0 = 0; k0 < D_MODEL; k0 += BK) {
    // A tile: 128x32 bf16 = 512 x 16B loads, 2 per thread.
#pragma unroll
    for (int i = 0; i < 2; ++i) {
      const int f = tid + 256 * i;
      const int r = f >> 2;
      const int c = (f & 3) << 3;
      *(v8bf*)(As + r * LDT + c) =
          *(const v8bf*)(Ctx + (size_t)(m0 + r) * D_MODEL + k0 + c);
    }
    // B tile: Wo 32x64 f32 -> bf16 transposed.
#pragma unroll
    for (int i = 0; i < 2; ++i) {
      const int f = tid + 256 * i;
      const int kr = f >> 4;
      const int nc = (f & 15) << 2;
      v4f val = *(const v4f*)(W + (size_t)(k0 + kr) * D_MODEL + n0 + nc);
      Bs[(nc + 0) * LDT + kr] = (bf16)val.x;
      Bs[(nc + 1) * LDT + kr] = (bf16)val.y;
      Bs[(nc + 2) * LDT + kr] = (bf16)val.z;
      Bs[(nc + 3) * LDT + kr] = (bf16)val.w;
    }
    __syncthreads();

    v16bf a = load_a_frag(As + (16 * wv + l16) * LDT, lane);
#pragma unroll
    for (int t = 0; t < 4; ++t) {
      v16bf b = load_b_frag(Bs + (16 * t + l16) * LDT, lane);
      acc[t] = wmma_bf16(a, b, acc[t]);
    }
    __syncthreads();
  }

#pragma unroll
  for (int t = 0; t < 4; ++t) {
    const int n = n0 + 16 * t + l16;
    const float bv = bias[n];
#pragma unroll
    for (int r = 0; r < 8; ++r) {
      const int m = m0 + 16 * wv + r + half * 8;
      Y[(size_t)m * D_MODEL + n] = acc[t][r] + bv;
    }
  }
}

// ---------------------------------------------------------------------------
extern "C" void kernel_launch(void* const* d_in, const int* in_sizes, int n_in,
                              void* d_out, int out_size, void* d_ws,
                              size_t ws_size, hipStream_t stream) {
  const float* q  = (const float*)d_in[0];
  const float* k  = (const float*)d_in[1];
  const float* v  = (const float*)d_in[2];
  const float* Wq = (const float*)d_in[3];
  const float* Wk = (const float*)d_in[4];
  const float* Wv = (const float*)d_in[5];
  const float* Wo = (const float*)d_in[6];
  const float* bq = (const float*)d_in[7];
  const float* bk = (const float*)d_in[8];
  const float* bv = (const float*)d_in[9];
  const float* bo = (const float*)d_in[10];
  float* out = (float*)d_out;

  const size_t elems = (size_t)MTOT * D_MODEL;     // 8M bf16 per buffer
  if (ws_size < 4 * elems * sizeof(bf16)) return;  // 64 MB workspace needed

  bf16* Qh  = (bf16*)d_ws;
  bf16* Kh  = Qh + elems;
  bf16* Vh  = Kh + elems;   // transposed layout [B,H,HD,S]
  bf16* Ctx = Vh + elems;

  const float qscale = 0.125f;  // 1/sqrt(HD=64), folded into Q projection

  dim3 gGemm(D_MODEL / BN, MTOT / BM);  // (16, 64)
  proj_gemm_kernel<<<gGemm, 256, 0, stream>>>(q, Wq, bq, Qh, qscale, 0);
  proj_gemm_kernel<<<gGemm, 256, 0, stream>>>(k, Wk, bk, Kh, 1.0f, 0);
  proj_gemm_kernel<<<gGemm, 256, 0, stream>>>(v, Wv, bv, Vh, 1.0f, 1);

  dim3 gAttn(SEQ / 64, NHEAD, BATCH);   // (32, 16, 4)
  attn_kernel<<<gAttn, 128, 0, stream>>>(Qh, Kh, Vh, Ctx);

  out_gemm_kernel<<<gGemm, 256, 0, stream>>>(Ctx, Wo, bo, out);
}
